// Attention_70909910057750
// MI455X (gfx1250) — matrix operand order repaired and checked by reading, not compiled, not run on hardware
//
#include <hip/hip_runtime.h>
#include <hip/hip_bf16.h>
#include <math.h>
#include <stdint.h>

#define DEVINL __device__ __forceinline__

typedef __attribute__((ext_vector_type(16))) _Float16 v16h;
typedef __attribute__((ext_vector_type(8)))  _Float16 v8h;
typedef __attribute__((ext_vector_type(8)))  float    v8f;
typedef unsigned int u32x4 __attribute__((ext_vector_type(4)));
typedef int          i32x8 __attribute__((ext_vector_type(8)));
typedef int          i32x4 __attribute__((ext_vector_type(4)));

static constexpr int  Bsz  = 2;
static constexpr int  Tseq = 2048;
static constexpr int  Cdim = 2048;
static constexpr int  NH   = 16;    // query heads
static constexpr int  KVH  = 4;     // kv heads (GQA group = 4)
static constexpr int  HD   = 128;   // head dim
static constexpr float SOFTCAP = 50.0f;

DEVINL v8f wmma_f16(v16h a, v16h b, v8f c) {
  // emits v_wmma_f32_16x16x32_f16
  return __builtin_amdgcn_wmma_f32_16x16x32_f16(
      false, a, false, b, (short)0, c, false, false);
}

// ---------------------------------------------------------------------------
// Stage a 32(K) x 16(N) f32 weight tile into LDS (row-major f32).
// TDM path (one tensor_load_to_lds per block, wave 0) when the builtin
// exists; otherwise cooperative coalesced float4 loads + ds_store_b128.
// Caller must __syncthreads() after this returns.
// ---------------------------------------------------------------------------
DEVINL void stage_b_tile(float* ldsB, const float* __restrict__ W, int ldw,
                         int kk, int coloff, int tid, int wave) {
#if defined(__AMDGCN__) && __has_builtin(__builtin_amdgcn_tensor_load_to_lds)
  if (wave == 0) {
    const uint64_t ga =
        (uint64_t)(uintptr_t)(const void*)(W + (long)kk * ldw + coloff);
    const unsigned lds_off = (unsigned)(uintptr_t)(void*)ldsB; // flat[31:0]=LDS offset
    u32x4 g0;
    g0[0] = 1u;                                  // count=1, no gather
    g0[1] = lds_off;                             // lds_addr
    g0[2] = (unsigned)(ga & 0xffffffffu);        // global_addr[31:0]
    g0[3] = (unsigned)((ga >> 32) & 0x1ffffffu)  // global_addr[56:32]
          | (2u << 30);                          // type = 2 ("image")
    i32x8 g1;
    g1[0] = (2 << 16);            // data_size = 4 bytes
    g1[1] = (int)(16u << 16);     // tensor_dim0 = 16 (bits 79:48 -> w1[31:16])
    g1[2] = (int)(32u << 16);     // tensor_dim1 = 32 (bits 111:80 -> w2[31:16])
    g1[3] = (int)(16u << 16);     // tile_dim0 = 16   (bits 127:112)
    g1[4] = 32;                   // tile_dim1 = 32, tile_dim2 = 0
    g1[5] = ldw;                  // tensor_dim0_stride[31:0]
    g1[6] = 0;                    // stride hi / dim1_stride lo
    g1[7] = 0;
    i32x4 gz4 = {0, 0, 0, 0};
    i32x8 gz8 = {0, 0, 0, 0, 0, 0, 0, 0};
    // 6-arg form (clang-23 / therock-10.0 headers)
    __builtin_amdgcn_tensor_load_to_lds(g0, g1, gz4, gz4, gz8, 0);
    __builtin_amdgcn_s_wait_tensorcnt(0);
  }
#else
  // cooperative: 128 threads x float4 = 512 floats = full 32x16 tile
  const int e = tid * 4;
  const int r = e >> 4;
  const int c = e & 15;
  const float4 v = *reinterpret_cast<const float4*>(
      W + (long)(kk + r) * ldw + coloff + c);
  *reinterpret_cast<float4*>(ldsB + r * 16 + c) = v;
#endif
}

// B-fragment (32x16) read out of the staged f32 LDS tile.
DEVINL v16h read_b_frag(const float* ldsB, int lane) {
  const float* lb = ldsB + (lane & 15);
  const int bkb = (lane & 16) ? 8 : 0;
  v16h b;
#pragma unroll
  for (int i = 0; i < 16; ++i) b[i] = (_Float16)lb[(bkb + i + (i & 8)) * 16];
  return b;
}

// ---------------------------------------------------------------------------
// Kernel 1: QKV projection. One wave computes one 16x16 output tile; the
// 4 waves of a block share one staged weight tile (same blockIdx.y).
// Writes Q -> [B][NH][T][HD], K/V -> [B][KVH][T][HD] as f16.
// ---------------------------------------------------------------------------
__global__ void __launch_bounds__(128)
qkv_proj_kernel(const float* __restrict__ x,
                const float* __restrict__ qw,
                const float* __restrict__ kw,
                const float* __restrict__ vw,
                _Float16* __restrict__ Qr,
                _Float16* __restrict__ Kr,
                _Float16* __restrict__ Vr) {
  __shared__ float ldsB[32 * 16];
  const int tid  = threadIdx.x;
  const int lane = tid & 31;
  const int wave = tid >> 5;
  const int mt   = blockIdx.x * 4 + wave;   // 0..255
  const int ct   = blockIdx.y;              // 0..191 (Q:128, K:32, V:32)
  const int m0   = mt * 16;

  const float* W; int ldw, coloff, which;
  if (ct < 128)      { W = qw; ldw = NH * HD;  coloff = ct * 16;         which = 0; }
  else if (ct < 160) { W = kw; ldw = KVH * HD; coloff = (ct - 128) * 16; which = 1; }
  else               { W = vw; ldw = KVH * HD; coloff = (ct - 160) * 16; which = 2; }

  const int arow = m0 + (lane & 15);
  const int akb  = (lane & 16) ? 8 : 0;
  const float4* ap4 = reinterpret_cast<const float4*>(x + (long)arow * Cdim + akb);

  v8f acc = {0.f, 0.f, 0.f, 0.f, 0.f, 0.f, 0.f, 0.f};
  for (int kk = 0; kk < Cdim; kk += 32) {
    stage_b_tile(ldsB, W, ldw, kk, coloff, tid, wave);
    __syncthreads();

    // A fragment: two contiguous 32B chunks -> 4x global_load_b128
    const int q4 = kk >> 2;
    float4 a0 = ap4[q4 + 0];
    float4 a1 = ap4[q4 + 1];
    float4 a2 = ap4[q4 + 4];
    float4 a3 = ap4[q4 + 5];
    v16h a;
    a[0]=(_Float16)a0.x; a[1]=(_Float16)a0.y; a[2]=(_Float16)a0.z; a[3]=(_Float16)a0.w;
    a[4]=(_Float16)a1.x; a[5]=(_Float16)a1.y; a[6]=(_Float16)a1.z; a[7]=(_Float16)a1.w;
    a[8]=(_Float16)a2.x; a[9]=(_Float16)a2.y; a[10]=(_Float16)a2.z; a[11]=(_Float16)a2.w;
    a[12]=(_Float16)a3.x; a[13]=(_Float16)a3.y; a[14]=(_Float16)a3.z; a[15]=(_Float16)a3.w;

    v16h b = read_b_frag(ldsB, lane);
    acc = wmma_f16(a, b, acc);
    __syncthreads();
  }

  const int rb   = (lane & 16) ? 8 : 0;
  const int bcol = coloff + (lane & 15);
  const int head = bcol >> 7;
  const int h    = bcol & 127;
#pragma unroll
  for (int r = 0; r < 8; ++r) {
    const int row  = m0 + rb + r;
    const int bidx = row >> 11;
    const int t    = row & (Tseq - 1);
    _Float16 v = (_Float16)acc[r];
    if (which == 0)
      Qr[(((long)(bidx * NH + head)) * Tseq + t) * HD + h] = v;
    else if (which == 1)
      Kr[(((long)(bidx * KVH + head)) * Tseq + t) * HD + h] = v;
    else
      Vr[(((long)(bidx * KVH + head)) * Tseq + t) * HD + h] = v;
  }
}

// ---------------------------------------------------------------------------
// Kernel 2: in-place rotate-half RoPE; Q gets 1/sqrt(HD) folded in.
// ---------------------------------------------------------------------------
__global__ void __launch_bounds__(256)
rope_kernel(_Float16* __restrict__ X, long nrows, float scale) {
  long idx = (long)blockIdx.x * blockDim.x + threadIdx.x;
  if (idx >= nrows * 64) return;
  const int  j  = (int)(idx & 63);
  const long rr = idx >> 6;
  const int  t  = (int)(rr & (Tseq - 1));
  float inv = powf(10000.0f, -(float)j * (1.0f / 64.0f));
  float ang = (float)t * inv;
  float s, c;
  sincosf(ang, &s, &c);
  _Float16* p = X + rr * HD;
  float x0 = (float)p[j];
  float x1 = (float)p[j + 64];
  p[j]      = (_Float16)((x0 * c - x1 * s) * scale);
  p[j + 64] = (_Float16)((x1 * c + x0 * s) * scale);
}

// ---------------------------------------------------------------------------
// Kernel 3: fused causal flash attention with tanh softcap.
// ---------------------------------------------------------------------------
__global__ void __launch_bounds__(128)
flash_attn_kernel(const _Float16* __restrict__ Qr,
                  const _Float16* __restrict__ Kr,
                  const _Float16* __restrict__ Vr,
                  _Float16* __restrict__ enc) {
  __shared__ _Float16 Pbuf[4][16 * 32];
  const int lane = threadIdx.x & 31;
  const int wave = threadIdx.x >> 5;
  const int tile = blockIdx.x * 4 + wave;   // 0..4095
  const int mt   = tile & 127;
  const int n    = (tile >> 7) & 15;
  const int b    = tile >> 11;
  const int m0   = mt * 16;
  const int kvh  = n >> 2;

  const _Float16* Qp = Qr + (((long)(b * NH + n)) * Tseq + m0) * HD;
  const _Float16* Kp = Kr + ((long)(b * KVH + kvh)) * Tseq * HD;
  const _Float16* Vp = Vr + ((long)(b * KVH + kvh)) * Tseq * HD;

  // preload Q A-fragments (vector loads: two 16B chunks per 32-wide h chunk)
  v16h qf[4];
  {
    const int row = lane & 15;
    const int akb = (lane & 16) ? 16 : 0;
#pragma unroll
    for (int hc = 0; hc < 4; ++hc) {
      const _Float16* p = Qp + (long)row * HD + hc * 32 + akb;
      v8h lo = *(const v8h*)(p);
      v8h hi = *(const v8h*)(p + 8);
      qf[hc] = __builtin_shufflevector(lo, hi, 0,1,2,3,4,5,6,7,
                                               8,9,10,11,12,13,14,15);
    }
  }

  v8f acc[8];
#pragma unroll
  for (int i = 0; i < 8; ++i) acc[i] = (v8f){0.f,0.f,0.f,0.f,0.f,0.f,0.f,0.f};
  float mrow[8], lrow[8];
#pragma unroll
  for (int r = 0; r < 8; ++r) { mrow[r] = -1e30f; lrow[r] = 0.f; }

  const int rb   = (lane & 16) ? 8 : 0;
  const int colS = lane & 15;
  _Float16* Pls  = &Pbuf[wave][0];

  const int send = m0 + 16;
  for (int s0 = 0; s0 < send; s0 += 32) {
    v8f S0 = {0.f,0.f,0.f,0.f,0.f,0.f,0.f,0.f};
    v8f S1 = {0.f,0.f,0.f,0.f,0.f,0.f,0.f,0.f};
    {
      const int kb = (lane & 16) ? 16 : 0;
      const _Float16* kp0 = Kp + (long)(s0 + colS) * HD + kb;
      const _Float16* kp1 = kp0 + 16 * HD;
#pragma unroll
      for (int hc = 0; hc < 4; ++hc) {
        v16h b0 = *(const v16h*)(kp0 + hc * 32);
        v16h b1 = *(const v16h*)(kp1 + hc * 32);
        S0 = wmma_f16(qf[hc], b0, S0);
        S1 = wmma_f16(qf[hc], b1, S1);
      }
    }

    float corr[8];
#pragma unroll
    for (int r = 0; r < 8; ++r) {
      const int t = m0 + rb + r;
      float s0v = SOFTCAP * tanhf(S0[r] * (1.0f / SOFTCAP));
      float s1v = SOFTCAP * tanhf(S1[r] * (1.0f / SOFTCAP));
      if (s0 + colS > t)      s0v = -1e30f;
      if (s0 + 16 + colS > t) s1v = -1e30f;
      float vmax = fmaxf(s0v, s1v);
      vmax = fmaxf(vmax, __shfl_xor(vmax, 1, 32));
      vmax = fmaxf(vmax, __shfl_xor(vmax, 2, 32));
      vmax = fmaxf(vmax, __shfl_xor(vmax, 4, 32));
      vmax = fmaxf(vmax, __shfl_xor(vmax, 8, 32));
      float mn = fmaxf(mrow[r], vmax);
      float p0 = __expf(s0v - mn);
      float p1 = __expf(s1v - mn);
      float ps = p0 + p1;
      ps += __shfl_xor(ps, 1, 32);
      ps += __shfl_xor(ps, 2, 32);
      ps += __shfl_xor(ps, 4, 32);
      ps += __shfl_xor(ps, 8, 32);
      float cr = __expf(mrow[r] - mn);
      lrow[r]  = lrow[r] * cr + ps;
      mrow[r]  = mn;
      corr[r]  = cr;
      Pls[(rb + r) * 32 + colS]      = (_Float16)p0;
      Pls[(rb + r) * 32 + 16 + colS] = (_Float16)p1;
    }

    // same-wave LDS RAW: DS pipe is in-order; fence compiler + ds counter.
    __asm__ volatile("s_wait_dscnt 0" ::: "memory");

    v16h pa;
    {
      const _Float16* p = Pls + (long)(lane & 15) * 32 + ((lane & 16) ? 16 : 0);
      v8h lo = *(const v8h*)(p);
      v8h hi = *(const v8h*)(p + 8);
      pa = __builtin_shufflevector(lo, hi, 0,1,2,3,4,5,6,7,
                                           8,9,10,11,12,13,14,15);
    }

#pragma unroll
    for (int hc = 0; hc < 8; ++hc) {
#pragma unroll
      for (int r = 0; r < 8; ++r) acc[hc][r] *= corr[r];
      v16h vb;
      const int vkb = (lane & 16) ? 16 : 0;
      const _Float16* vp = Vp + (long)(s0 + vkb) * HD + hc * 16 + (lane & 15);
#pragma unroll
      for (int i = 0; i < 16; ++i) vb[i] = vp[(long)i * HD];
      acc[hc] = wmma_f16(pa, vb, acc[hc]);
    }
  }

#pragma unroll
  for (int r = 0; r < 8; ++r) {
    float inv = 1.0f / lrow[r];
    const long row = (long)b * Tseq + m0 + rb + r;
    _Float16* op = enc + row * (NH * HD) + n * HD + (lane & 15);
#pragma unroll
    for (int hc = 0; hc < 8; ++hc)
      op[hc * 16] = (_Float16)(acc[hc][r] * inv);
  }
}

// ---------------------------------------------------------------------------
// Kernel 4: output projection  encoded(f16) @ out_kernel(f32) -> f32.
// 4 waves of a block share one staged weight tile (same column tile).
// ---------------------------------------------------------------------------
__global__ void __launch_bounds__(128)
out_proj_kernel(const _Float16* __restrict__ enc,
                const float* __restrict__ ow,
                float* __restrict__ out) {
  __shared__ float ldsB[32 * 16];
  const int tid  = threadIdx.x;
  const int lane = tid & 31;
  const int wave = tid >> 5;
  const int nt   = blockIdx.x & 127;              // col tile: shared by block
  const int mt   = (blockIdx.x >> 7) * 4 + wave;  // 0..255
  const int m0   = mt * 16, n0 = nt * 16;

  const int arow = m0 + (lane & 15);
  const int akb  = (lane & 16) ? 8 : 0;
  const _Float16* ap = enc + (long)arow * (NH * HD) + akb;

  v8f acc = {0.f,0.f,0.f,0.f,0.f,0.f,0.f,0.f};
  for (int kk = 0; kk < NH * HD; kk += 32) {
    stage_b_tile(ldsB, ow, Cdim, kk, n0, tid, wave);
    __syncthreads();

    v8h lo = *(const v8h*)(ap + kk);
    v8h hi = *(const v8h*)(ap + kk + 16);
    v16h a = __builtin_shufflevector(lo, hi, 0,1,2,3,4,5,6,7,
                                             8,9,10,11,12,13,14,15);
    v16h b = read_b_frag(ldsB, lane);
    acc = wmma_f16(a, b, acc);
    __syncthreads();
  }
  const int rb = (lane & 16) ? 8 : 0;
  const int bcol = n0 + (lane & 15);
#pragma unroll
  for (int r = 0; r < 8; ++r)
    out[(long)(m0 + rb + r) * Cdim + bcol] = acc[r];
}

// ---------------------------------------------------------------------------
extern "C" void kernel_launch(void* const* d_in, const int* in_sizes, int n_in,
                              void* d_out, int out_size, void* d_ws, size_t ws_size,
                              hipStream_t stream) {
  const float* x  = (const float*)d_in[0];
  const float* qw = (const float*)d_in[1];
  const float* kw = (const float*)d_in[2];
  const float* vw = (const float*)d_in[3];
  const float* ow = (const float*)d_in[4];
  float* out = (float*)d_out;

  // workspace (f16): Q 16MB | K 4MB | V 4MB | encoded 16MB = 40MB
  _Float16* Qr  = (_Float16*)d_ws;
  _Float16* Kr  = Qr + (long)Bsz * NH  * Tseq * HD;
  _Float16* Vr  = Kr + (long)Bsz * KVH * Tseq * HD;
  _Float16* enc = Vr + (long)Bsz * KVH * Tseq * HD;

  qkv_proj_kernel<<<dim3(64, 192), dim3(128), 0, stream>>>(x, qw, kw, vw, Qr, Kr, Vr);

  const long qrows = (long)Bsz * NH * Tseq;
  rope_kernel<<<(unsigned)((qrows * 64 + 255) / 256), 256, 0, stream>>>(
      Qr, qrows, 0.08838834764831845f);
  const long krows = (long)Bsz * KVH * Tseq;
  rope_kernel<<<(unsigned)((krows * 64 + 255) / 256), 256, 0, stream>>>(
      Kr, krows, 1.0f);

  flash_attn_kernel<<<1024, 128, 0, stream>>>(Qr, Kr, Vr, enc);

  // 256 row-tiles x 128 col-tiles, 4 row-tiles per block
  out_proj_kernel<<<64 * 128, 128, 0, stream>>>(enc, ow, out);
}
